// MANDOGraphClassifier_43233140802047
// MI455X (gfx1250) — matrix operand order repaired and checked by reading, not compiled, not run on hardware
//
#include <hip/hip_runtime.h>
#include <hip/hip_bf16.h>
#include <math.h>

// ---- Problem constants (from reference) ----
constexpr int kN   = 50000;   // nodes (= 3125 * 16, exact M tiles)
constexpr int kP   = 4;       // metapaths
constexpr int kE   = 500000;  // edges per metapath
constexpr int kIN  = 128;     // input feature dim (K)
constexpr int kH   = 8;       // heads
constexpr int kF   = 32;      // feats per head
constexpr int kHF  = 256;     // H*F (Nc)
constexpr int kHS  = 128;     // semantic hidden
constexpr int kB   = 128;     // graphs
constexpr int kOUT = 2;       // classes

typedef __attribute__((ext_vector_type(16))) _Float16 v16h;
typedef __attribute__((ext_vector_type(8)))  _Float16 v8h;
typedef __attribute__((ext_vector_type(8)))  float    v8f;

union HFrag { v16h v; v8h p[2]; _Float16 h[16]; };

// ---------- f32 -> f16 conversion, 8 elems / thread ----------
__global__ __launch_bounds__(256) void cvt_f16x8_kernel(
    const float* __restrict__ in, _Float16* __restrict__ out, int n8) {
  int i = blockIdx.x * blockDim.x + threadIdx.x;
  if (i >= n8) return;
  const float4* in4 = (const float4*)in;
  float4 x = in4[2 * i], y = in4[2 * i + 1];
  v8h o;
  o[0] = (_Float16)x.x; o[1] = (_Float16)x.y; o[2] = (_Float16)x.z; o[3] = (_Float16)x.w;
  o[4] = (_Float16)y.x; o[5] = (_Float16)y.y; o[6] = (_Float16)y.z; o[7] = (_Float16)y.w;
  *(v8h*)(out + (size_t)i * 8) = o;
}

__global__ void fill_f32_kernel(float* __restrict__ p, float v, int n) {
  int i = blockIdx.x * blockDim.x + threadIdx.x;
  if (i < n) p[i] = v;
}

// ---------- pack Wg[p] (f32 [K,Nc] row-major) into WMMA B-fragment order ----------
// Bp layout: [(p*16 + ntile)*4 + kblk][lane][16 halves], lane's 16 halves contiguous.
// lane (half = lane>>4, nr = lane&15) holds B rows kblk*32 + half*16 + (0..15), col ntile*16+nr.
__global__ __launch_bounds__(256) void pack_B_kernel(
    const float* __restrict__ Wg, _Float16* __restrict__ Bp) {
  int t = blockIdx.x * blockDim.x + threadIdx.x;   // kP*16*4*32 = 8192 threads
  if (t >= kP * 16 * 4 * 32) return;
  int lane  = t & 31;
  int kblk  = (t >> 5) & 3;
  int ntile = (t >> 7) & 15;
  int p     = t >> 11;
  int half  = lane >> 4;
  int n     = ntile * 16 + (lane & 15);
  int k0    = kblk * 32 + half * 16;
  const float* src = Wg + (size_t)p * kIN * kHF;
  _Float16* outp = Bp + (size_t)t * 16;
#pragma unroll
  for (int i = 0; i < 16; ++i)
    outp[i] = (_Float16)src[(size_t)(k0 + i) * kHF + n];
}

// ---------- WMMA GEMM: C[M,256] = A[M,128] * B, fully unrolled, no guards ----------
// Block = 128 threads = 4 waves; each wave computes a 16x32 slab (2 N-tiles),
// reusing its A fragment.  Grid: (M/16, 2).
__global__ __launch_bounds__(128) void gemm_f16_wmma_kernel(
    const _Float16* __restrict__ A,    // [M,128] row-major f16
    const _Float16* __restrict__ Bp,   // packed fragments (one metapath)
    float* __restrict__ C)             // [M,256] f32
{
  const int lane  = threadIdx.x & 31;
  const int wave  = threadIdx.x >> 5;
  const int m0    = blockIdx.x * 16;
  const int tile0 = blockIdx.y * 8 + wave * 2;   // global N-tile index (0..15)
  const int half  = lane >> 4;
  const int nr    = lane & 15;
  const _Float16* arow = A + (size_t)(m0 + (lane & 15)) * kIN;

  v8f acc0 = {}, acc1 = {};
#pragma unroll
  for (int kblk = 0; kblk < 4; ++kblk) {
    const int k0 = kblk * 32;
    HFrag a, b0, b1;
    // A fragment: 16 contiguous halves per lane, k = 16*half + s (matches packed B)
    a.p[0] = *(const v8h*)(arow + k0 + 16 * half);
    a.p[1] = *(const v8h*)(arow + k0 + 16 * half + 8);
    // B fragments: pre-packed, 32B contiguous per lane
    const _Float16* bp0 = Bp + (((size_t)tile0 * 4 + kblk) * 32 + lane) * 16;
    const _Float16* bp1 = Bp + (((size_t)(tile0 + 1) * 4 + kblk) * 32 + lane) * 16;
    b0.p[0] = *(const v8h*)bp0; b0.p[1] = *(const v8h*)(bp0 + 8);
    b1.p[0] = *(const v8h*)bp1; b1.p[1] = *(const v8h*)(bp1 + 8);
    acc0 = __builtin_amdgcn_wmma_f32_16x16x32_f16(false, a.v, false, b0.v,
                                                  (short)0, acc0, false, false);
    acc1 = __builtin_amdgcn_wmma_f32_16x16x32_f16(false, a.v, false, b1.v,
                                                  (short)0, acc1, false, false);
  }
  // C/D layout: VGPR r -> (m = m0 + r + 8*half, n = lane&15)
#pragma unroll
  for (int r = 0; r < 8; ++r) {
    size_t m = (size_t)(m0 + r + 8 * half);
    C[m * kHF + tile0 * 16 + nr]       = acc0[r];
    C[m * kHF + (tile0 + 1) * 16 + nr] = acc1[r];
  }
}

// ---------- GAT attention dot products: el/er [N,H] ----------
__global__ __launch_bounds__(256) void attn_dots_kernel(
    const float* __restrict__ hp, const float* __restrict__ al,
    const float* __restrict__ ar, float* __restrict__ el, float* __restrict__ er)
{
  int t = blockIdx.x * blockDim.x + threadIdx.x;
  if (t >= kN * kH) return;
  int i = t >> 3, hh = t & 7;
  const float4* row4 = (const float4*)(hp + (size_t)i * kHF + hh * kF);
  const float4* pl4  = (const float4*)(al + hh * kF);
  const float4* pr4  = (const float4*)(ar + hh * kF);
  float sl = 0.f, sr = 0.f;
#pragma unroll
  for (int q = 0; q < kF / 4; ++q) {
    float4 x = row4[q], l = pl4[q], r = pr4[q];
    sl += x.x * l.x + x.y * l.y + x.z * l.z + x.w * l.w;
    sr += x.x * r.x + x.y * r.y + x.z * r.z + x.w * r.w;
  }
  el[t] = sl; er[t] = sr;
}

__device__ __forceinline__ void atomicMaxFloat(float* addr, float value) {
  if (value >= 0.f)
    atomicMax((int*)addr, __float_as_int(value));
  else
    atomicMin((unsigned int*)addr, __float_as_uint(value));
}

// ---------- edge pass 1: e = leaky_relu(el[s]+er[d]); per-dst max ----------
__global__ __launch_bounds__(256) void edge_scores_kernel(
    const int* __restrict__ src, const int* __restrict__ dst,
    const float* __restrict__ el, const float* __restrict__ er,
    float* __restrict__ ebuf, float* __restrict__ mmax)
{
  int t = blockIdx.x * blockDim.x + threadIdx.x;
  if (t >= kE * kH) return;
  int e = t >> 3, hh = t & 7;
  int s = src[e], d = dst[e];
  float x = el[s * kH + hh] + er[d * kH + hh];
  x = (x > 0.f) ? x : 0.2f * x;   // leaky_relu, slope 0.2
  ebuf[t] = x;
  atomicMaxFloat(&mmax[d * kH + hh], x);
}

// ---------- edge pass 2: a = exp(e - m[d]); den[d] += a ----------
__global__ __launch_bounds__(256) void edge_exp_den_kernel(
    const int* __restrict__ dst, float* __restrict__ ebuf,
    const float* __restrict__ mmax, float* __restrict__ den)
{
  int t = blockIdx.x * blockDim.x + threadIdx.x;
  if (t >= kE * kH) return;
  int e = t >> 3, hh = t & 7;
  int d = dst[e];
  float a = expf(ebuf[t] - mmax[d * kH + hh]);
  ebuf[t] = a;
  atomicAdd(&den[d * kH + hh], a);
}

// ---------- edge pass 3: z[d] += (a/den[d]) * hp[s]  (one wave per edge) ----------
__global__ __launch_bounds__(256) void edge_aggregate_kernel(
    const int* __restrict__ src, const int* __restrict__ dst,
    const float* __restrict__ ebuf, const float* __restrict__ den,
    const float* __restrict__ hp, float* __restrict__ zp)
{
  int e = blockIdx.x * 8 + (threadIdx.x >> 5);
  if (e >= kE) return;
  int lane = threadIdx.x & 31;
  int s = src[e], d = dst[e];
  const float* hrow = hp + (size_t)s * kHF;
  float* orow = zp + (size_t)d * kHF;
#pragma unroll
  for (int it = 0; it < kH; ++it) {        // head `it` <-> features it*32..it*32+31
    float alpha = ebuf[(size_t)e * kH + it] / den[(size_t)d * kH + it];
    int f = it * 32 + lane;
    atomicAdd(&orow[f], alpha * hrow[f]);
  }
}

// ---------- z = elu(out + bias) ----------
__global__ __launch_bounds__(256) void z_finalize_kernel(
    float* __restrict__ zp, const float* __restrict__ bias)
{
  int t = blockIdx.x * blockDim.x + threadIdx.x;
  if (t >= kN * kHF) return;
  float x = zp[t] + bias[t & (kHF - 1)];
  zp[t] = (x > 0.f) ? x : (expf(x) - 1.f);
}

// ---------- semantic attention score per (node, metapath) ----------
__global__ __launch_bounds__(128) void sem_w_kernel(
    const float* __restrict__ z, const float* __restrict__ W1,
    const float* __restrict__ b1, const float* __restrict__ W2,
    float* __restrict__ wbuf)
{
  __shared__ float zrow[kHF];
  __shared__ float red[kHS];
  int i = blockIdx.x, p = blockIdx.y, j = threadIdx.x;
  const float* zr = z + ((size_t)p * kN + i) * kHF;
  zrow[j] = zr[j];
  zrow[j + kHS] = zr[j + kHS];
  __syncthreads();
  float acc = b1[j];
  for (int k = 0; k < kHF; ++k) acc += zrow[k] * W1[k * kHS + j];
  red[j] = tanhf(acc) * W2[j];
  __syncthreads();
  for (int s2 = kHS / 2; s2 > 0; s2 >>= 1) {
    if (j < s2) red[j] += red[j + s2];
    __syncthreads();
  }
  if (j == 0) wbuf[(size_t)p * kN + i] = red[0];
}

// ---------- sum of w over nodes, per metapath ----------
__global__ __launch_bounds__(256) void wsum_kernel(const float* __restrict__ wbuf,
                                                   float* __restrict__ wsum)
{
  __shared__ float red[256];
  int p = blockIdx.y;
  float s = 0.f;
  for (int i = blockIdx.x * blockDim.x + threadIdx.x; i < kN; i += gridDim.x * blockDim.x)
    s += wbuf[(size_t)p * kN + i];
  red[threadIdx.x] = s; __syncthreads();
  for (int s2 = 128; s2 > 0; s2 >>= 1) {
    if ((int)threadIdx.x < s2) red[threadIdx.x] += red[threadIdx.x + s2];
    __syncthreads();
  }
  if (threadIdx.x == 0) atomicAdd(&wsum[p], red[0]);
}

__global__ void beta_kernel(const float* __restrict__ wsum, float* __restrict__ beta) {
  float v[kP]; float mx = -3.0e38f;
  for (int p = 0; p < kP; ++p) { v[p] = wsum[p] / (float)kN; mx = fmaxf(mx, v[p]); }
  float s = 0.f;
  for (int p = 0; p < kP; ++p) { v[p] = expf(v[p] - mx); s += v[p]; }
  for (int p = 0; p < kP; ++p) beta[p] = v[p] / s;
}

// ---------- fused = sum_p beta_p * z_p ; per-graph segment sums ----------
__global__ __launch_bounds__(256) void fused_pool_kernel(
    const float* __restrict__ z, const float* __restrict__ beta,
    const int* __restrict__ gid, float* __restrict__ seg, float* __restrict__ cnt)
{
  int i = blockIdx.x * 8 + (threadIdx.x >> 5);
  if (i >= kN) return;
  int lane = threadIdx.x & 31;
  int g = gid[i];
  float b0 = beta[0], b1 = beta[1], b2 = beta[2], b3 = beta[3];
  const size_t stride = (size_t)kN * kHF;
#pragma unroll
  for (int it = 0; it < 8; ++it) {
    int f = it * 32 + lane;
    size_t base = (size_t)i * kHF + f;
    float v = b0 * z[base] + b1 * z[stride + base] + b2 * z[2 * stride + base] + b3 * z[3 * stride + base];
    atomicAdd(&seg[(size_t)g * kHF + f], v);
  }
  if (lane == 0) atomicAdd(&cnt[g], 1.0f);
}

__global__ __launch_bounds__(256) void pooled_kernel(
    const float* __restrict__ seg, const float* __restrict__ cnt, float* __restrict__ pooled)
{
  int t = blockIdx.x * blockDim.x + threadIdx.x;
  if (t >= kB * kHF) return;
  pooled[t] = seg[t] / fmaxf(cnt[t >> 8], 1.0f);
}

__global__ __launch_bounds__(256) void logits_kernel(
    const float* __restrict__ pooled, const float* __restrict__ W,
    const float* __restrict__ b, float* __restrict__ logits)
{
  int t = blockIdx.x * blockDim.x + threadIdx.x;
  if (t >= kB * kOUT) return;
  int bb = t / kOUT, o = t % kOUT;
  float s = b[o];
  for (int k = 0; k < kHF; ++k) s += pooled[(size_t)bb * kHF + k] * W[k * kOUT + o];
  logits[t] = s;
}

// ---------------------------------------------------------------------------
extern "C" void kernel_launch(void* const* d_in, const int* in_sizes, int n_in,
                              void* d_out, int out_size, void* d_ws, size_t ws_size,
                              hipStream_t stream) {
  (void)in_sizes; (void)n_in; (void)out_size; (void)ws_size;
  const float* h        = (const float*)d_in[0];
  const float* Wg       = (const float*)d_in[1];
  const float* attn_l   = (const float*)d_in[2];
  const float* attn_r   = (const float*)d_in[3];
  const float* gat_bias = (const float*)d_in[4];
  const float* sem_W1   = (const float*)d_in[5];
  const float* sem_b1   = (const float*)d_in[6];
  const float* sem_W2   = (const float*)d_in[7];
  const float* cls_W    = (const float*)d_in[8];
  const float* cls_b    = (const float*)d_in[9];
  const int*   src      = (const int*)d_in[10];
  const int*   dst      = (const int*)d_in[11];
  const int*   gid      = (const int*)d_in[12];
  float* out = (float*)d_out;   // [0,256): logits ; [256, 256+32768): pooled

  // ---- workspace layout ----
  char* ws = (char*)d_ws;
  size_t off = 0;
  auto alloc = [&](size_t bytes) -> char* {
    char* p = ws + off;
    off += (bytes + 255) & ~(size_t)255;
    return p;
  };
  _Float16* h16  = (_Float16*)alloc((size_t)kN * kIN * sizeof(_Float16));
  _Float16* Bp   = (_Float16*)alloc((size_t)kP * 16 * 4 * 32 * 16 * sizeof(_Float16));
  float* hp   = (float*)alloc((size_t)kN * kHF * sizeof(float));          // per-p scratch
  float* z    = (float*)alloc((size_t)kP * kN * kHF * sizeof(float));     // [P,N,HF]
  float* el   = (float*)alloc((size_t)kN * kH * sizeof(float));
  float* er   = (float*)alloc((size_t)kN * kH * sizeof(float));
  float* mmax = (float*)alloc((size_t)kN * kH * sizeof(float));
  float* den  = (float*)alloc((size_t)kN * kH * sizeof(float));
  float* ebuf = (float*)alloc((size_t)kE * kH * sizeof(float));
  float* wbuf = (float*)alloc((size_t)kP * kN * sizeof(float));
  float* wsum = (float*)alloc(kP * sizeof(float));
  float* beta = (float*)alloc(kP * sizeof(float));
  float* seg  = (float*)alloc((size_t)kB * kHF * sizeof(float));
  float* cnt  = (float*)alloc((size_t)kB * sizeof(float));

  // ---- GEMM operand prep: A -> f16, W -> packed f16 fragments ----
  {
    int n8 = (kN * kIN) / 8;
    cvt_f16x8_kernel<<<(n8 + 255) / 256, 256, 0, stream>>>(h, h16, n8);
    pack_B_kernel<<<(kP * 16 * 4 * 32 + 255) / 256, 256, 0, stream>>>(Wg, Bp);
  }
  hipMemsetAsync(z, 0, (size_t)kP * kN * kHF * sizeof(float), stream);
  hipMemsetAsync(wsum, 0, kP * sizeof(float), stream);
  hipMemsetAsync(seg, 0, (size_t)kB * kHF * sizeof(float), stream);
  hipMemsetAsync(cnt, 0, (size_t)kB * sizeof(float), stream);

  // ---- per-metapath GATConv (sequential so hp + z_p stay L2-resident) ----
  for (int p = 0; p < kP; ++p) {
    const int*   srcp = src + (size_t)p * kE;
    const int*   dstp = dst + (size_t)p * kE;
    float*       zp   = z + (size_t)p * kN * kHF;

    // hp = h @ Wg[p]   (WMMA f16 in / f32 accumulate)
    gemm_f16_wmma_kernel<<<dim3(kN / 16, 2), 128, 0, stream>>>(
        h16, Bp + (size_t)p * 16 * 4 * 32 * 16, hp);

    // el/er + softmax-state init
    {
      int n = kN * kH;
      attn_dots_kernel<<<(n + 255) / 256, 256, 0, stream>>>(
          hp, attn_l + (size_t)p * kH * kF, attn_r + (size_t)p * kH * kF, el, er);
      fill_f32_kernel<<<(n + 255) / 256, 256, 0, stream>>>(mmax, -3.0e38f, n);
    }
    hipMemsetAsync(den, 0, (size_t)kN * kH * sizeof(float), stream);

    // edge softmax (3 streaming passes) + aggregation
    {
      int nt = kE * kH;
      edge_scores_kernel<<<(nt + 255) / 256, 256, 0, stream>>>(srcp, dstp, el, er, ebuf, mmax);
      edge_exp_den_kernel<<<(nt + 255) / 256, 256, 0, stream>>>(dstp, ebuf, mmax, den);
      edge_aggregate_kernel<<<(kE + 7) / 8, 256, 0, stream>>>(srcp, dstp, ebuf, den, hp, zp);
    }
    // elu(out + bias)
    {
      int n = kN * kHF;
      z_finalize_kernel<<<(n + 255) / 256, 256, 0, stream>>>(zp, gat_bias + (size_t)p * kHF);
    }
  }

  // ---- semantic attention over metapaths ----
  sem_w_kernel<<<dim3(kN, kP), 128, 0, stream>>>(z, sem_W1, sem_b1, sem_W2, wbuf);
  wsum_kernel<<<dim3(64, kP), 256, 0, stream>>>(wbuf, wsum);
  beta_kernel<<<1, 1, 0, stream>>>(wsum, beta);

  // ---- fuse + per-graph mean pooling + classifier ----
  fused_pool_kernel<<<(kN + 7) / 8, 256, 0, stream>>>(z, beta, gid, seg, cnt);
  float* pooled_out = out + kB * kOUT;
  {
    int n = kB * kHF;
    pooled_kernel<<<(n + 255) / 256, 256, 0, stream>>>(seg, cnt, pooled_out);
  }
  logits_kernel<<<1, 256, 0, stream>>>(pooled_out, cls_W, cls_b, out);
}
